// ParallelAutoencoder_82463372083236
// MI455X (gfx1250) — matrix operand order repaired and checked by reading, not compiled, not run on hardware
//
#include <hip/hip_runtime.h>

// Problem constants (reference: T=4096, H=1024, L=2)
#define T_SEQ 4096
#define HID   1024
#define FOURH 4096
#define NLAY  2

// Persistent recurrence config: 256 WGs, each owns H/256 = 4 hidden elements
// -> 16 gate rows of W_hh (4 gates x 4) = 16*1024 f16 = 32KB in LDS.
#define RGRID  256
#define RSLICE (HID / RGRID)      // 4
#define RROWS  (4 * RSLICE)       // 16

typedef _Float16 h8  __attribute__((ext_vector_type(8)));
typedef _Float16 h16 __attribute__((ext_vector_type(16)));
typedef float    f8  __attribute__((ext_vector_type(8)));

__device__ __forceinline__ float sigmoidf(float x) { return 1.0f / (1.0f + __expf(-x)); }

// ---------------------------------------------------------------- utilities
__global__ __launch_bounds__(256) void k_f32_to_f16(const float* __restrict__ s,
                                                    _Float16* __restrict__ d, int n) {
  int i = blockIdx.x * 256 + threadIdx.x;
  if (i < n) d[i] = (_Float16)s[i];
}

__global__ __launch_bounds__(256) void k_fill_f32(float* __restrict__ d, float v, int n) {
  int i = blockIdx.x * 256 + threadIdx.x;
  if (i < n) d[i] = v;
}

__global__ __launch_bounds__(256) void k_copy_f32(float* __restrict__ d,
                                                  const float* __restrict__ s, int n) {
  int i = blockIdx.x * 256 + threadIdx.x;
  if (i < n) d[i] = s[i];
}

__global__ __launch_bounds__(256) void k_init_flags(int* __restrict__ f, int n, int g0) {
  int i = blockIdx.x * 256 + threadIdx.x;
  if (i < n) f[i] = (i == 0) ? g0 : 0;
}

// ---------------------------------------------------------------- WMMA GEMM
// C[M,N] = A[M,K] * B[N,K]^T (+ epilogue).  A,B f16 row-major, K%32==0.
// Block = 256 threads = 8 waves (wave32), tiled 2(M) x 4(N); per-wave 32x64.
// Workgroup tile: 64(M) x 256(N).

// A fragment 16x32 (MxK), ISA layout: lanes0-15 M=lane, K{0..7,16..23};
// lanes16-31 M=lane-16, K{8..15,24..31}.
__device__ __forceinline__ h16 load_a_frag(const _Float16* __restrict__ A, int K,
                                           int m0, int k0, int lane) {
  int l16 = lane & 15;
  int hi8 = (lane >> 4) << 3;  // 0 or 8
  const _Float16* p = A + (size_t)(m0 + l16) * K + (k0 + hi8);
  h8 lo = *(const h8*)p;
  h8 hh = *(const h8*)(p + 16);
  return __builtin_shufflevector(lo, hh, 0, 1, 2, 3, 4, 5, 6, 7, 8, 9, 10, 11, 12, 13, 14, 15);
}

// B fragment 32x16 (KxN): lanes0-15 N=lane, K{0..7,16..23}; lanes16-31 N=lane-16, K{8..15,24..31}.
__device__ __forceinline__ h16 load_b_frag(const _Float16* __restrict__ B, int K,
                                           int n0, int k0, int lane) {
  int l16  = lane & 15;
  int hi16 = (lane >> 4) << 3;  // 0 or 8
  const _Float16* p = B + (size_t)(n0 + l16) * K + (k0 + hi16);
  h8 lo = *(const h8*)p;
  h8 hh = *(const h8*)(p + 16);
  return __builtin_shufflevector(lo, hh, 0, 1, 2, 3, 4, 5, 6, 7, 8, 9, 10, 11, 12, 13, 14, 15);
}

// xg epilogue: C = acc + b_ih[n] + b_hh[n]
__global__ __launch_bounds__(256) void k_gemm_xg(const _Float16* __restrict__ A,
                                                 const _Float16* __restrict__ B,
                                                 const float* __restrict__ bi,
                                                 const float* __restrict__ bh,
                                                 float* __restrict__ C,
                                                 int M, int N, int K) {
  int lane = threadIdx.x & 31;
  int wave = threadIdx.x >> 5;
  int m_w  = blockIdx.y * 64 + (wave & 1) * 32;
  int n_w  = blockIdx.x * 256 + (wave >> 1) * 64;
  f8 acc[2][4] = {};
  for (int k0 = 0; k0 < K; k0 += 32) {
    h16 a0 = load_a_frag(A, K, m_w, k0, lane);
    h16 a1 = load_a_frag(A, K, m_w + 16, k0, lane);
#pragma unroll
    for (int j = 0; j < 4; ++j) {
      h16 b = load_b_frag(B, K, n_w + 16 * j, k0, lane);
      acc[0][j] = __builtin_amdgcn_wmma_f32_16x16x32_f16(false, a0, false, b, (short)0,
                                                         acc[0][j], false, false);
      acc[1][j] = __builtin_amdgcn_wmma_f32_16x16x32_f16(false, a1, false, b, (short)0,
                                                         acc[1][j], false, false);
    }
  }
  int l16 = lane & 15, mhi = (lane >> 4) << 3;
#pragma unroll
  for (int i = 0; i < 2; ++i) {
#pragma unroll
    for (int j = 0; j < 4; ++j) {
      int n = n_w + 16 * j + l16;
      float bias = bi[n] + bh[n];
      int mb = m_w + 16 * i + mhi;
#pragma unroll
      for (int v = 0; v < 8; ++v)
        C[(size_t)(mb + v) * N + n] = acc[i][j][v] + bias;
    }
  }
}

// output epilogue: leaky-relu(acc + ob[n]) stored time-flipped into D
__global__ __launch_bounds__(256) void k_gemm_out(const _Float16* __restrict__ A,
                                                  const _Float16* __restrict__ B,
                                                  const float* __restrict__ ob,
                                                  float* __restrict__ D,
                                                  int M, int N, int K) {
  int lane = threadIdx.x & 31;
  int wave = threadIdx.x >> 5;
  int m_w  = blockIdx.y * 64 + (wave & 1) * 32;
  int n_w  = blockIdx.x * 256 + (wave >> 1) * 64;
  f8 acc[2][4] = {};
  for (int k0 = 0; k0 < K; k0 += 32) {
    h16 a0 = load_a_frag(A, K, m_w, k0, lane);
    h16 a1 = load_a_frag(A, K, m_w + 16, k0, lane);
#pragma unroll
    for (int j = 0; j < 4; ++j) {
      h16 b = load_b_frag(B, K, n_w + 16 * j, k0, lane);
      acc[0][j] = __builtin_amdgcn_wmma_f32_16x16x32_f16(false, a0, false, b, (short)0,
                                                         acc[0][j], false, false);
      acc[1][j] = __builtin_amdgcn_wmma_f32_16x16x32_f16(false, a1, false, b, (short)0,
                                                         acc[1][j], false, false);
    }
  }
  int l16 = lane & 15, mhi = (lane >> 4) << 3;
#pragma unroll
  for (int i = 0; i < 2; ++i) {
#pragma unroll
    for (int j = 0; j < 4; ++j) {
      int n = n_w + 16 * j + l16;
      float bias = ob[n];
      int mb = m_w + 16 * i + mhi;
#pragma unroll
      for (int v = 0; v < 8; ++v) {
        int m = mb + v;
        float val = acc[i][j][v] + bias;
        val = (val > 0.0f) ? val : 0.01f * val;
        D[(size_t)(M - 1 - m) * N + n] = val;  // time flip
      }
    }
  }
}

// ------------------------------------------------- constant x-gate GEMV (dec layer 0)
// out[r] = dot(W[r,:], v) + bi[r] + bh[r], W f16 [4H,H]. 8 rows / 256-thread WG.
__global__ __launch_bounds__(256) void k_gemv_xgc(const _Float16* __restrict__ W,
                                                  const float* __restrict__ v,
                                                  const float* __restrict__ bi,
                                                  const float* __restrict__ bh,
                                                  float* __restrict__ out) {
  int row  = blockIdx.x * 8 + (threadIdx.x >> 5);
  int lane = threadIdx.x & 31;
  const _Float16* wr = W + (size_t)row * HID;
  float s = 0.0f;
  for (int k = lane; k < HID; k += 32) s += (float)wr[k] * v[k];
#pragma unroll
  for (int off = 16; off > 0; off >>= 1) s += __shfl_down(s, off, 32);
  if (lane == 0) out[row] = s + bi[row] + bh[row];
}

// ---------------------------------------------------------------- persistent LSTM recurrence
// 256 co-resident WGs; WG `wg` owns h[wg*4 .. wg*4+3] and the matching 16 rows of
// W_hh (i/f/g/o), pinned in LDS for the whole sequence. Per step: wait flags[t]==RGRID,
// stage h_t (4KB) in LDS, 16 dot products, gate math, publish h slice to hs[t+1],
// release-increment flags[t+1]. Write-once flags => no reset races; graph-replay safe
// because k_init_flags re-zeros before each phase.
__global__ __launch_bounds__(256) void k_lstm_recurrence(const _Float16* __restrict__ Whh,
                                                         const float* __restrict__ xg,
                                                         int xg_stride,
                                                         float* __restrict__ hs,
                                                         const float* __restrict__ c0,
                                                         float* __restrict__ cT,
                                                         int* __restrict__ flags) {
  __shared__ _Float16 Wl[RROWS * HID];   // 32 KB
  __shared__ float    hb[HID];           // 4 KB
  __shared__ float    red[RROWS][17];    // padded reduce buffer

  const int wg  = blockIdx.x;
  const int tid = threadIdx.x;

  // Stage this WG's 16 gate rows of W_hh into LDS (16B vector copies).
  for (int c = tid; c < RROWS * (HID / 8); c += 256) {
    int r  = c >> 7;          // local row 0..15
    int ch = c & 127;         // 8-elem chunk
    int q  = r / RSLICE;      // gate 0..3
    int j  = r % RSLICE;      // 0..3
    const h8* src = (const h8*)(Whh + (size_t)(q * HID + wg * RSLICE + j) * HID) + ch;
    ((h8*)(Wl + r * HID))[ch] = *src;
  }
  float c_state = (tid < RSLICE) ? c0[wg * RSLICE + tid] : 0.0f;
  __syncthreads();

  const int rowl = tid >> 4;  // 0..15: local gate row
  const int part = tid & 15;  // 0..15: 64-element partial

  for (int t = 0; t < T_SEQ; ++t) {
    if (tid == 0) {
      while (__hip_atomic_load(&flags[t], __ATOMIC_ACQUIRE, __HIP_MEMORY_SCOPE_AGENT) < RGRID)
        __builtin_amdgcn_s_sleep(1);
    }
    __syncthreads();

    const float* hsrc = hs + (size_t)t * HID;
    for (int k = tid; k < HID; k += 256) hb[k] = hsrc[k];
    __syncthreads();

    float s = 0.0f;
    const _Float16* wr = Wl + rowl * HID + part * 64;
    const float* hp = hb + part * 64;
#pragma unroll 8
    for (int k = 0; k < 64; ++k) s += (float)wr[k] * hp[k];
    red[rowl][part] = s;
    __syncthreads();

    if (tid < RSLICE) {
      const float* xr = xg + (size_t)t * xg_stride;
      float g4[4];
#pragma unroll
      for (int q = 0; q < 4; ++q) {
        float a = xr[q * HID + wg * RSLICE + tid];
#pragma unroll
        for (int p = 0; p < 16; ++p) a += red[q * RSLICE + tid][p];
        g4[q] = a;
      }
      float iv = sigmoidf(g4[0]);
      float fv = sigmoidf(g4[1]);
      float gv = tanhf(g4[2]);
      float ov = sigmoidf(g4[3]);
      c_state = fv * c_state + iv * gv;
      float h = ov * tanhf(c_state);
      hs[(size_t)(t + 1) * HID + wg * RSLICE + tid] = h;
      if (t == T_SEQ - 1) cT[wg * RSLICE + tid] = c_state;
      __threadfence();  // make h slice visible before flag release
    }
    __syncthreads();
    if (tid == 0)
      __hip_atomic_fetch_add(&flags[t + 1], 1, __ATOMIC_RELEASE, __HIP_MEMORY_SCOPE_AGENT);
  }
}

// ---------------------------------------------------------------- host orchestration
extern "C" void kernel_launch(void* const* d_in, const int* in_sizes, int n_in,
                              void* d_out_v, int out_size, void* d_ws, size_t ws_size,
                              hipStream_t stream) {
  const float* x    = (const float*)d_in[0];
  const float* eWih = (const float*)d_in[1];
  const float* eWhh = (const float*)d_in[2];
  const float* ebih = (const float*)d_in[3];
  const float* ebhh = (const float*)d_in[4];
  const float* dWih = (const float*)d_in[5];
  const float* dWhh = (const float*)d_in[6];
  const float* dbih = (const float*)d_in[7];
  const float* dbhh = (const float*)d_in[8];
  const float* oW   = (const float*)d_in[9];
  const float* ob   = (const float*)d_in[10];
  float* out = (float*)d_out_v;

  size_t off = 0;
  auto alloc = [&](size_t bytes) -> char* {
    char* p = (char*)d_ws + off;
    off += (bytes + 255) & ~(size_t)255;
    return p;
  };
  const size_t WELEM = (size_t)NLAY * FOURH * HID;   // per stacked weight tensor
  _Float16* w_eih = (_Float16*)alloc(WELEM * 2);
  _Float16* w_ehh = (_Float16*)alloc(WELEM * 2);
  _Float16* w_dih = (_Float16*)alloc(WELEM * 2);
  _Float16* w_dhh = (_Float16*)alloc(WELEM * 2);
  _Float16* w_out = (_Float16*)alloc((size_t)HID * HID * 2);
  _Float16* x_h   = (_Float16*)alloc((size_t)T_SEQ * HID * 2);
  _Float16* hsh   = (_Float16*)alloc((size_t)T_SEQ * HID * 2);      // reused f16 hs staging
  float* xg    = (float*)alloc((size_t)T_SEQ * FOURH * 4);          // reused x-gates
  float* hs_e0 = (float*)alloc((size_t)(T_SEQ + 1) * HID * 4);
  float* hs_e1 = (float*)alloc((size_t)(T_SEQ + 1) * HID * 4);
  float* hs_d0 = (float*)alloc((size_t)(T_SEQ + 1) * HID * 4);
  float* hs_d1 = (float*)alloc((size_t)(T_SEQ + 1) * HID * 4);
  float* xgc   = (float*)alloc((size_t)FOURH * 4);                  // constant dec0 gates
  float* zc    = (float*)alloc((size_t)HID * 4);                    // zero c0
  float* cfin  = (float*)alloc((size_t)4 * HID * 4);                // cT: enc0,enc1,dec0,dec1
  int*   flags = (int*)alloc((size_t)(T_SEQ + 1) * 4);

  auto conv = [&](const float* s, _Float16* d, size_t n) {
    k_f32_to_f16<<<dim3((unsigned)((n + 255) / 256)), dim3(256), 0, stream>>>(s, d, (int)n);
  };
  auto fill = [&](float* d, float v, size_t n) {
    k_fill_f32<<<dim3((unsigned)((n + 255) / 256)), dim3(256), 0, stream>>>(d, v, (int)n);
  };
  auto copy = [&](float* d, const float* s, size_t n) {
    k_copy_f32<<<dim3((unsigned)((n + 255) / 256)), dim3(256), 0, stream>>>(d, s, (int)n);
  };
  auto flags_init = [&]() {
    k_init_flags<<<dim3((T_SEQ + 1 + 255) / 256), dim3(256), 0, stream>>>(flags, T_SEQ + 1, RGRID);
  };
  auto gemm_xg = [&](const _Float16* A, const _Float16* B, const float* bi, const float* bh) {
    k_gemm_xg<<<dim3(FOURH / 256, T_SEQ / 64), dim3(256), 0, stream>>>(A, B, bi, bh, xg,
                                                                       T_SEQ, FOURH, HID);
  };
  auto recur = [&](const _Float16* Whh, const float* g, int gs, float* hs,
                   const float* c0p, float* cTp) {
    flags_init();
    k_lstm_recurrence<<<dim3(RGRID), dim3(256), 0, stream>>>(Whh, g, gs, hs, c0p, cTp, flags);
  };

  // f32 -> f16 staging of weights and input
  conv(x, x_h, (size_t)T_SEQ * HID);
  conv(eWih, w_eih, WELEM);
  conv(eWhh, w_ehh, WELEM);
  conv(dWih, w_dih, WELEM);
  conv(dWhh, w_dhh, WELEM);
  conv(oW, w_out, (size_t)HID * HID);
  fill(zc, 0.0f, HID);
  fill(hs_e0, 0.0f, HID);  // enc h0 = 0
  fill(hs_e1, 0.0f, HID);

  // ---- encoder layer 0
  gemm_xg(x_h, w_eih, ebih, ebhh);
  recur(w_ehh, xg, FOURH, hs_e0, zc, cfin + 0 * HID);

  // ---- encoder layer 1
  conv(hs_e0 + HID, hsh, (size_t)T_SEQ * HID);
  gemm_xg(hsh, w_eih + WELEM / 2, ebih + FOURH, ebhh + FOURH);
  recur(w_ehh + WELEM / 2, xg, FOURH, hs_e1, zc, cfin + 1 * HID);

  const float* final_hidden = hs_e1 + (size_t)T_SEQ * HID;
  copy(out + (size_t)T_SEQ * HID, final_hidden, HID);  // second output

  // ---- decoder layer 0 (constant input = final_hidden -> constant x-gates)
  k_gemv_xgc<<<dim3(FOURH / 8), dim3(256), 0, stream>>>(w_dih, final_hidden, dbih, dbhh, xgc);
  copy(hs_d0, hs_e0 + (size_t)T_SEQ * HID, HID);       // h0 = enc_h[0]
  recur(w_dhh, xgc, 0, hs_d0, cfin + 0 * HID, cfin + 2 * HID);

  // ---- decoder layer 1
  conv(hs_d0 + HID, hsh, (size_t)T_SEQ * HID);
  gemm_xg(hsh, w_dih + WELEM / 2, dbih + FOURH, dbhh + FOURH);
  copy(hs_d1, hs_e1 + (size_t)T_SEQ * HID, HID);       // h0 = enc_h[1]
  recur(w_dhh + WELEM / 2, xg, FOURH, hs_d1, cfin + 1 * HID, cfin + 3 * HID);

  // ---- output projection + leaky relu + flip, fused into d_out
  conv(hs_d1 + HID, hsh, (size_t)T_SEQ * HID);
  k_gemm_out<<<dim3(HID / 256, T_SEQ / 64), dim3(256), 0, stream>>>(hsh, w_out, ob, out,
                                                                    T_SEQ, HID, HID);
}
